// DiscQtDecoderHead_9440338116880
// MI455X (gfx1250) — matrix operand order repaired and checked by reading, not compiled, not run on hardware
//
#include <hip/hip_runtime.h>

// ---------------- problem constants ----------------
#define T_STEPS 20
#define HIDN    512
#define NROWS   6400      // B*K = 64*100
#define EMBD    300
#define KA0     320       // padded embedding width (mult of 32)
#define KPAD0   832       // 320 + 512   (26 chunks of 32, even)
#define KA1     512
#define KPAD1   1024      // 512 + 512   (32 chunks of 32, even)
#define NANS    30000

typedef __attribute__((ext_vector_type(16))) _Float16 v16h;
typedef __attribute__((ext_vector_type(8)))  _Float16 v8h;
typedef __attribute__((ext_vector_type(8)))  float    v8f;

__device__ __forceinline__ float sigmoidf_(float x) { return 1.0f / (1.0f + __expf(-x)); }

// ---------------- weight prep: combined f16 [4H][Kpad] + fused bias ----------------
__global__ void prep_weights(const float* __restrict__ wih0, const float* __restrict__ whh0,
                             const float* __restrict__ bih0, const float* __restrict__ bhh0,
                             const float* __restrict__ wih1, const float* __restrict__ whh1,
                             const float* __restrict__ bih1, const float* __restrict__ bhh1,
                             _Float16* __restrict__ W0c, _Float16* __restrict__ W1c,
                             float* __restrict__ bias0, float* __restrict__ bias1)
{
    const int idx = blockIdx.x * blockDim.x + threadIdx.x;
    const int n0 = 4 * HIDN * KPAD0;
    const int n1 = 4 * HIDN * KPAD1;
    if (idx < n0) {
        const int r = idx / KPAD0, c = idx % KPAD0;
        float v = 0.0f;
        if (c < EMBD)       v = wih0[r * EMBD + c];
        else if (c >= KA0)  v = whh0[r * HIDN + (c - KA0)];
        W0c[idx] = (_Float16)v;
    }
    if (idx < n1) {
        const int r = idx / KPAD1, c = idx % KPAD1;
        const float v = (c < KA1) ? wih1[r * HIDN + c] : whh1[r * HIDN + (c - KA1)];
        W1c[idx] = (_Float16)v;
    }
    if (idx < 4 * HIDN) {
        bias0[idx] = bih0[idx] + bhh0[idx];
        bias1[idx] = bih1[idx] + bhh1[idx];
    }
}

// ---------------- zero h (ping buffer) and c for both layers ----------------
__global__ void init_state(_Float16* __restrict__ h0, _Float16* __restrict__ h1,
                           float* __restrict__ c0, float* __restrict__ c1)
{
    const int idx = blockIdx.x * blockDim.x + threadIdx.x;
    if (idx < NROWS * HIDN) {
        h0[idx] = (_Float16)0.0f;
        h1[idx] = (_Float16)0.0f;
        c0[idx] = 0.0f;
        c1[idx] = 0.0f;
    }
}

// ---------------- one-shot embedding: Xemb[t][n][KA0] f16 (padded, masked) ----------------
// one thread = 8 contiguous k; grid covers T*NROWS*KA0/8 threads
__global__ void embed_kernel(const float* __restrict__ Wemb,
                             const int*   __restrict__ opt,     // [NROWS*T]
                             _Float16*    __restrict__ Xemb)    // [T*NROWS*KA0]
{
    const int idx = blockIdx.x * blockDim.x + threadIdx.x;
    const int total = T_STEPS * NROWS * (KA0 / 8);
    if (idx >= total) return;
    const int k0 = (idx % (KA0 / 8)) * 8;
    const int nt = idx / (KA0 / 8);          // = t*NROWS + n
    const int t  = nt / NROWS;
    const int n  = nt % NROWS;
    const int tok = opt[n * T_STEPS + t];

    v8h out = (v8h)((_Float16)0.0f);
    if (tok != 0 && k0 < EMBD) {
        const float* p = Wemb + (size_t)tok * EMBD + k0;
        if (k0 + 8 <= EMBD) {
#pragma unroll
            for (int q = 0; q < 8; ++q) out[q] = (_Float16)p[q];
        } else {
#pragma unroll
            for (int q = 0; q < 8; ++q)
                if (k0 + q < EMBD) out[q] = (_Float16)p[q];
        }
    }
    *(v8h*)&Xemb[(size_t)nt * KA0 + k0] = out;
}

// ---------------- one fused LSTM time step (WMMA GEMM + cell update) ----------------
// grid = (NROWS/64, HIDN/64), block = 256 (8 waves of 32)
#define LDA 40    // f16 stride of LDS A tile (64 x 32, padded)
#define GS  264   // f16 stride of LDS gates tile (64 x 256, padded)

__global__ __launch_bounds__(256)
void lstm_step(const _Float16* __restrict__ Xsrc,     // x input, stride KA f16 per row
               const _Float16* __restrict__ Hin,      // previous h (f16)
               _Float16*       __restrict__ Hout,     // new h (f16, other buffer)
               float*          __restrict__ Cst,      // c state (f32, in place)
               const _Float16* __restrict__ Wc,       // combined weights [4H][Kpad] f16
               const float*    __restrict__ bias,     // fused bias [4H]
               const int*      __restrict__ lens,     // [NROWS]
               int t, int KA, int Kpad)
{
    __shared__ __align__(32) _Float16 ldsA[2][64 * LDA];
    __shared__ __align__(32) _Float16 ldsG[64 * GS];

    const int tid     = threadIdx.x;
    const int lane    = tid & 31;
    const int wid     = tid >> 5;
    const int nbase   = blockIdx.x * 64;
    const int colbase = blockIdx.y * 64;
    const int r0  = (wid & 3) * 16;   // wave's row sub-tile
    const int j0  = (wid >> 2) * 128; // wave's gate-column sub-tile (of 256)
    const int m16 = lane & 15;
    const int hi  = lane >> 4;

    // cooperative A-tile loader mapping: thread -> (row, 8-wide col chunk)
    const int ar = tid >> 2;          // 0..63
    const int ac = (tid & 3) << 3;    // 0,8,16,24
    const int an = nbase + ar;
    const _Float16* xrow = Xsrc + (size_t)an * KA;
    const _Float16* hrow = Hin  + (size_t)an * HIDN - KA;  // indexed by k0 >= KA

    // per-sub-tile B row pointers (invariant over K)
    const _Float16* bptr[8];
#pragma unroll
    for (int i = 0; i < 8; ++i) {
        const int jj   = j0 + i * 16;
        const int g    = jj >> 6;
        const int wrow = g * HIDN + colbase + (jj & 63) + m16;
        bptr[i] = Wc + (size_t)wrow * Kpad + hi * 16;
    }

    // A-tile fetch: single 16B load either from X slab or from Hin
    auto loadA = [&](int kc) -> v8h {
        const int k0 = (kc << 5) + ac;
        const _Float16* src = (k0 < KA) ? (xrow + k0) : (hrow + k0);
        return *(const v8h*)src;
    };
    // A fragment from LDS (16x32 f16): lane m, K pattern e -> hi*16 + e
    auto afragFrom = [&](const _Float16* buf) -> v16h {
        const v8h alo = *(const v8h*)&buf[(r0 + m16) * LDA + hi * 16];
        const v8h ahi = *(const v8h*)&buf[(r0 + m16) * LDA + hi * 16 + 8];
        v16h a;
#pragma unroll
        for (int e = 0; e < 8; ++e) { a[e] = alo[e]; a[e + 8] = ahi[e]; }
        return a;
    };

    v8f acc[8];
#pragma unroll
    for (int i = 0; i < 8; ++i) acc[i] = (v8f)(0.0f);

    const int nchunks = Kpad >> 5;    // 26 or 32 (always even)

    // ---- prologue: stage chunk 0 into ldsA[0], preload its B fragments ----
    {
        const v8h a0 = loadA(0);
        *(v8h*)&ldsA[0][ar * LDA + ac] = a0;
    }
    v16h ba[8], bb[8];
#pragma unroll
    for (int i = 0; i < 8; ++i) ba[i] = *(const v16h*)(bptr[i]);
    __syncthreads();

    // ---- main loop, 2 chunks per iteration (ping-pong LDS + B registers) ----
    for (int kc = 0; kc < nchunks; kc += 2) {
        // even chunk kc: compute with ldsA[0]/ba, prefetch chunk kc+1
        {
            const v8h an1 = loadA(kc + 1);
#pragma unroll
            for (int i = 0; i < 8; ++i)
                bb[i] = *(const v16h*)(bptr[i] + (size_t)(kc + 1) * 32);
            const v16h afrag = afragFrom(ldsA[0]);
#pragma unroll
            for (int i = 0; i < 8; ++i)
                acc[i] = __builtin_amdgcn_wmma_f32_16x16x32_f16(
                    false, afrag, false, ba[i], (short)0, acc[i], false, false);
            *(v8h*)&ldsA[1][ar * LDA + ac] = an1;
            __syncthreads();
        }
        // odd chunk kc+1: compute with ldsA[1]/bb, prefetch chunk kc+2
        {
            const bool more = (kc + 2) < nchunks;
            v8h an2;
            if (more) {
                an2 = loadA(kc + 2);
#pragma unroll
                for (int i = 0; i < 8; ++i)
                    ba[i] = *(const v16h*)(bptr[i] + (size_t)(kc + 2) * 32);
            }
            const v16h afrag = afragFrom(ldsA[1]);
#pragma unroll
            for (int i = 0; i < 8; ++i)
                acc[i] = __builtin_amdgcn_wmma_f32_16x16x32_f16(
                    false, afrag, false, bb[i], (short)0, acc[i], false, false);
            if (more) {
                *(v8h*)&ldsA[0][ar * LDA + ac] = an2;
            }
            __syncthreads();
        }
    }

    // spill gate accumulators (f16) to LDS for cross-wave gate exchange
#pragma unroll
    for (int i = 0; i < 8; ++i) {
        const int col = j0 + i * 16 + m16;
#pragma unroll
        for (int v = 0; v < 8; ++v) {
            const int mm = r0 + v + hi * 8;   // D layout: lanes 0-15 M=v, lanes 16-31 M=v+8
            ldsG[mm * GS + col] = (_Float16)acc[i][v];
        }
    }
    __syncthreads();

    // fused LSTM cell: 64 rows x 64 h-cols, 16 outputs/thread
#pragma unroll 4
    for (int it = 0; it < 16; ++it) {
        const int idx = tid + it * 256;
        const int m   = idx >> 6;
        const int hc  = idx & 63;
        const int n   = nbase + m;
        const int col = colbase + hc;
        const float iv = (float)ldsG[m * GS + hc]        + bias[col];
        const float fv = (float)ldsG[m * GS + 64 + hc]   + bias[HIDN + col];
        const float gv = (float)ldsG[m * GS + 128 + hc]  + bias[2 * HIDN + col];
        const float ov = (float)ldsG[m * GS + 192 + hc]  + bias[3 * HIDN + col];
        const size_t off = (size_t)n * HIDN + col;
        const float c_old = Cst[off];
        const float c_new = sigmoidf_(fv) * c_old + sigmoidf_(iv) * tanhf(gv);
        const float h_new = sigmoidf_(ov) * tanhf(c_new);
        if (t < lens[n]) {
            Cst[off]  = c_new;
            Hout[off] = (_Float16)h_new;
        } else {
            Hout[off] = Hin[off];   // freeze past sequence end
        }
    }
}

// ---------------- final scores + relevance gather ----------------
__global__ void finalize_scores(const _Float16* __restrict__ H2,
                                const float* __restrict__ enc,
                                const int* __restrict__ qt_idx,
                                const int* __restrict__ opt_idx,
                                const unsigned char* __restrict__ rel,
                                float* __restrict__ out)
{
    const int n = blockIdx.x * blockDim.x + threadIdx.x;
    if (n >= NROWS) return;
    const int b = n / 100;
    const _Float16* h = H2 + (size_t)n * HIDN;
    const float* e = enc + (size_t)b * HIDN;
    float s = 0.0f;
    for (int k = 0; k < HIDN; k += 8) {
        const v8h hv = *(const v8h*)(h + k);
#pragma unroll
        for (int q = 0; q < 8; ++q) s += (float)hv[q] * e[k + q];
    }
    out[n] = s;
    out[NROWS + n] = rel[(size_t)qt_idx[b] * NANS + opt_idx[n]] ? 1.0f : 0.0f;
}

// ---------------- host launcher ----------------
extern "C" void kernel_launch(void* const* d_in, const int* in_sizes, int n_in,
                              void* d_out, int out_size, void* d_ws, size_t ws_size,
                              hipStream_t stream)
{
    (void)in_sizes; (void)n_in; (void)out_size; (void)ws_size;

    const float* enc     = (const float*)d_in[0];
    const int*   opt     = (const int*)d_in[1];
    const int*   opt_len = (const int*)d_in[2];
    const int*   qt_idx  = (const int*)d_in[3];
    const int*   opt_idx = (const int*)d_in[4];
    const float* Wemb    = (const float*)d_in[5];
    const float* wih0    = (const float*)d_in[6];
    const float* whh0    = (const float*)d_in[7];
    const float* bih0    = (const float*)d_in[8];
    const float* bhh0    = (const float*)d_in[9];
    const float* wih1    = (const float*)d_in[10];
    const float* whh1    = (const float*)d_in[11];
    const float* bih1    = (const float*)d_in[12];
    const float* bhh1    = (const float*)d_in[13];
    const unsigned char* rel = (const unsigned char*)d_in[14];

    // workspace carving (256B aligned)
    char* ws = (char*)d_ws;
    size_t off = 0;
    auto carve = [&](size_t bytes) -> char* {
        char* p = ws + off;
        off += (bytes + 255) & ~(size_t)255;
        return p;
    };
    _Float16* W0c   = (_Float16*)carve((size_t)4 * HIDN * KPAD0 * 2);
    _Float16* W1c   = (_Float16*)carve((size_t)4 * HIDN * KPAD1 * 2);
    float*    bias0 = (float*)   carve((size_t)4 * HIDN * 4);
    float*    bias1 = (float*)   carve((size_t)4 * HIDN * 4);
    _Float16* Xemb  = (_Float16*)carve((size_t)T_STEPS * NROWS * KA0 * 2);
    _Float16* h0buf[2], *h1buf[2];
    h0buf[0] = (_Float16*)carve((size_t)NROWS * HIDN * 2);
    h0buf[1] = (_Float16*)carve((size_t)NROWS * HIDN * 2);
    h1buf[0] = (_Float16*)carve((size_t)NROWS * HIDN * 2);
    h1buf[1] = (_Float16*)carve((size_t)NROWS * HIDN * 2);
    float* c0 = (float*)carve((size_t)NROWS * HIDN * 4);
    float* c1 = (float*)carve((size_t)NROWS * HIDN * 4);

    // 1) weights -> combined f16 + fused bias
    {
        const int n1 = 4 * HIDN * KPAD1;
        prep_weights<<<(n1 + 255) / 256, 256, 0, stream>>>(
            wih0, whh0, bih0, bhh0, wih1, whh1, bih1, bhh1, W0c, W1c, bias0, bias1);
    }
    // 2) zero state (workspace is poisoned; must re-init every call)
    {
        const int ne = NROWS * HIDN;
        init_state<<<(ne + 255) / 256, 256, 0, stream>>>(h0buf[0], h1buf[0], c0, c1);
    }
    // 3) one-shot masked embedding into padded f16 slab [t][n][KA0]
    {
        const int nth = T_STEPS * NROWS * (KA0 / 8);
        embed_kernel<<<(nth + 255) / 256, 256, 0, stream>>>(Wemb, opt, Xemb);
    }
    // 4) 20 recurrent steps, layer0 then layer1 (layer1 consumes layer0's fresh h)
    const dim3 grid(NROWS / 64, HIDN / 64);
    for (int t = 0; t < T_STEPS; ++t) {
        _Float16* h0o = h0buf[(t + 1) & 1];
        lstm_step<<<grid, 256, 0, stream>>>(
            Xemb + (size_t)t * NROWS * KA0, h0buf[t & 1], h0o, c0,
            W0c, bias0, opt_len, t, KA0, KPAD0);
        lstm_step<<<grid, 256, 0, stream>>>(
            h0o, h1buf[t & 1], h1buf[(t + 1) & 1], c1,
            W1c, bias1, opt_len, t, KA1, KPAD1);
    }
    // 5) scores + qt relevance (final h1 lives in buffer (19+1)&1 == 0)
    finalize_scores<<<(NROWS + 255) / 256, 256, 0, stream>>>(
        h1buf[0], enc, qt_idx, opt_idx, rel, (float*)d_out);
}
